// MixedAttention_19524921328248
// MI455X (gfx1250) — matrix operand order repaired and checked by reading, not compiled, not run on hardware
//
#include <hip/hip_runtime.h>
#include <hip/hip_bf16.h>
#include <math.h>
#include <stdint.h>

typedef __attribute__((ext_vector_type(16))) _Float16     v16h;
typedef __attribute__((ext_vector_type(8)))  _Float16     v8h;
typedef __attribute__((ext_vector_type(8)))  float        v8f;
typedef __attribute__((ext_vector_type(4)))  unsigned int v4u;
typedef __attribute__((ext_vector_type(8)))  unsigned int v8u;

#define Bsz   8
#define Ssz   1024
#define HIDc  768
#define RHc   384
#define NHc   6
#define HDc   64
#define Kc    9
#define MROWS (Bsz * Ssz)           // 8192
#define BPST  776                   // padded B-panel row stride in halves

// element j of the v16h A-operand maps to K-offset within a 32-wide K tile:
__device__ __forceinline__ int a_off(int hf, int j) {
    return hf * 16 + j;
}

// ---------------------------------------------------------------------------
// Weight convert (+optional transpose) to f16 [N,K] row-major.
// ---------------------------------------------------------------------------
__global__ void convert_wt(const float* __restrict__ W, _Float16* __restrict__ out,
                           int N, int K, int transposed)
{
    int i = blockIdx.x * blockDim.x + threadIdx.x;
    if (i >= N * K) return;
    int n = i / K, k = i - n * K;
    float x = transposed ? W[(size_t)k * N + n] : W[(size_t)n * K + k];
    out[i] = (_Float16)x;
}

// ---------------------------------------------------------------------------
// GEMM: C[M,N] = A[M,K] (f32) * B16[N,K] (f16) + bias; f32 accumulate.
// Optional f16 mirror C16 (same ldc). Block 128 = 4 waves, tile 64x64,
// each wave owns a 32x32 quadrant (2x2 fragments -> 4 WMMA per K-step).
//
// NGUARD==false (aligned) path: the full 64-row B panel (<=64x768 f16) is
// DMA'd into LDS once by the Tensor Data Mover (iterate mode: 1 row/iter,
// lds_addr_increment = 776 halves for a conflict-free padded stride), issued
// by wave 0 and completed with s_wait_tensorcnt. The K-loop then only stages
// the A tile. NGUARD==true keeps per-step staging with N guards (N=54 case).
// ---------------------------------------------------------------------------
template<bool NGUARD>
__global__ __launch_bounds__(128) void gemm_wmma(
    const float* __restrict__ A, int lda,
    const _Float16* __restrict__ B16, int ldb,
    const float* __restrict__ bias,
    float* __restrict__ C, int ldc,
    _Float16* __restrict__ C16,
    int M, int N, int Kdim)
{
    __shared__ _Float16 As[64][40];

    const int tid  = threadIdx.x;
    const int wave = tid >> 5;
    const int lane = tid & 31;
    const int wm   = wave >> 1;          // 0..1
    const int wn   = wave & 1;           // 0..1
    const int hf   = lane >> 4;
    const int l16  = lane & 15;

    const int m0 = blockIdx.y * 64;
    const int n0 = blockIdx.x * 64;

    const int lr = tid >> 1;             // 0..63 (row for loaders)
    const int lc = (tid & 1) * 16;       // 0 or 16

    v8f acc[2][2] = {{{}, {}}, {{}, {}}};

    if constexpr (!NGUARD) {
        __shared__ _Float16 Bp[64 * BPST];   // ~99 KB padded B panel

        if (wave == 0) {
            // ---- TDM descriptor (D#), ISA 8.3-8.6, iterate mode ----
            const uint64_t ga  = (uint64_t)(uintptr_t)(B16 + (size_t)n0 * ldb);
            const uint32_t lds = (uint32_t)(uintptr_t)(&Bp[0]);
            v4u g0 = { 1u,                                   // count=1, user D#
                       lds,                                  // lds_addr
                       (uint32_t)ga,                         // global_addr[31:0]
                       (uint32_t)((ga >> 32) & 0x01FFFFFFu)  // global_addr[56:32]
                       | 0x80000000u };                      // type=2 ("image")
            v8u g1 = { (1u << 16) | (1u << 19),              // data_size=2B, iterate
                       ((uint32_t)Kdim & 0xFFFFu) << 16,     // tensor_dim0[15:0]
                       64u << 16,                            // tensor_dim1 = 64
                       ((uint32_t)Kdim & 0xFFFFu) << 16,     // tile_dim0 = Kdim
                       1u,                                   // tile_dim1 = 1
                       (uint32_t)ldb,                        // tensor_dim0_stride
                       0u, 0u };
            v4u g2 = { 0u,
                       (uint32_t)BPST,                       // lds_addr_increment
                       (uint32_t)ldb,                        // global_addr_increment
                       63u << 16 };                          // iterate_count = 64 rows
            v4u g3 = { 0u, 0u, 0u, 0u };
            asm volatile("tensor_load_to_lds %0, %1, %2, %3"
                         :: "s"(g0), "s"(g1), "s"(g2), "s"(g3)
                         : "memory");
            __builtin_amdgcn_s_wait_tensorcnt(0);
        }
        __syncthreads();

        for (int k0 = 0; k0 < Kdim; k0 += 32) {
            // ---- stage A (f32 -> f16), 16 elements/thread, vectorized ----
            {
                const float* src = A + (size_t)(m0 + lr) * lda + k0 + lc;
                float4 f0 = *(const float4*)(src);
                float4 f1 = *(const float4*)(src + 4);
                float4 f2 = *(const float4*)(src + 8);
                float4 f3 = *(const float4*)(src + 12);
                v8h h0 = { (_Float16)f0.x, (_Float16)f0.y, (_Float16)f0.z, (_Float16)f0.w,
                           (_Float16)f1.x, (_Float16)f1.y, (_Float16)f1.z, (_Float16)f1.w };
                v8h h1 = { (_Float16)f2.x, (_Float16)f2.y, (_Float16)f2.z, (_Float16)f2.w,
                           (_Float16)f3.x, (_Float16)f3.y, (_Float16)f3.z, (_Float16)f3.w };
                *(v8h*)&As[lr][lc]     = h0;
                *(v8h*)&As[lr][lc + 8] = h1;
                if (k0 + 32 < Kdim) __builtin_prefetch(src + 32, 0, 1);
            }
            __syncthreads();

            v16h a[2], b[2];
            #pragma unroll
            for (int i = 0; i < 2; ++i)
                #pragma unroll
                for (int j = 0; j < 16; ++j)
                    a[i][j] = As[wm * 32 + i * 16 + l16][a_off(hf, j)];
            #pragma unroll
            for (int jf = 0; jf < 2; ++jf) {
                const _Float16* brow = &Bp[(size_t)(wn * 32 + jf * 16 + l16) * BPST
                                           + k0 + hf * 16];
                v8h lo = *(const v8h*)(brow);
                v8h hi = *(const v8h*)(brow + 8);
                #pragma unroll
                for (int j = 0; j < 8; ++j) { b[jf][j] = lo[j]; b[jf][8 + j] = hi[j]; }
            }
            #pragma unroll
            for (int i = 0; i < 2; ++i)
                #pragma unroll
                for (int jf = 0; jf < 2; ++jf)
                    acc[i][jf] = __builtin_amdgcn_wmma_f32_16x16x32_f16(
                        false, a[i], false, b[jf], (short)0, acc[i][jf], false, false);
            __syncthreads();
        }
    } else {
        __shared__ _Float16 Bs[64][40];

        for (int k0 = 0; k0 < Kdim; k0 += 32) {
            {
                const float* src = A + (size_t)(m0 + lr) * lda + k0 + lc;
                float4 f0 = *(const float4*)(src);
                float4 f1 = *(const float4*)(src + 4);
                float4 f2 = *(const float4*)(src + 8);
                float4 f3 = *(const float4*)(src + 12);
                v8h h0 = { (_Float16)f0.x, (_Float16)f0.y, (_Float16)f0.z, (_Float16)f0.w,
                           (_Float16)f1.x, (_Float16)f1.y, (_Float16)f1.z, (_Float16)f1.w };
                v8h h1 = { (_Float16)f2.x, (_Float16)f2.y, (_Float16)f2.z, (_Float16)f2.w,
                           (_Float16)f3.x, (_Float16)f3.y, (_Float16)f3.z, (_Float16)f3.w };
                *(v8h*)&As[lr][lc]     = h0;
                *(v8h*)&As[lr][lc + 8] = h1;
            }
            {
                v8h h0 = {}, h1 = {};
                if ((n0 + lr) < N) {
                    const _Float16* src = B16 + (size_t)(n0 + lr) * ldb + k0 + lc;
                    h0 = *(const v8h*)(src);
                    h1 = *(const v8h*)(src + 8);
                }
                *(v8h*)&Bs[lr][lc]     = h0;
                *(v8h*)&Bs[lr][lc + 8] = h1;
            }
            __syncthreads();

            v16h a[2], b[2];
            #pragma unroll
            for (int i = 0; i < 2; ++i)
                #pragma unroll
                for (int j = 0; j < 16; ++j)
                    a[i][j] = As[wm * 32 + i * 16 + l16][a_off(hf, j)];
            #pragma unroll
            for (int i = 0; i < 2; ++i)
                #pragma unroll
                for (int j = 0; j < 16; ++j)
                    b[i][j] = Bs[wn * 32 + i * 16 + l16][hf * 16 + j];

            #pragma unroll
            for (int i = 0; i < 2; ++i)
                #pragma unroll
                for (int jf = 0; jf < 2; ++jf)
                    acc[i][jf] = __builtin_amdgcn_wmma_f32_16x16x32_f16(
                        false, a[i], false, b[jf], (short)0, acc[i][jf], false, false);
            __syncthreads();
        }
    }

    #pragma unroll
    for (int i = 0; i < 2; ++i)
        #pragma unroll
        for (int jf = 0; jf < 2; ++jf)
            #pragma unroll
            for (int r = 0; r < 8; ++r) {
                int m = m0 + wm * 32 + i * 16 + hf * 8 + r;
                int n = n0 + wn * 32 + jf * 16 + l16;
                if (!NGUARD || n < N) {
                    float val = acc[i][jf][r] + bias[n];
                    C[(size_t)m * ldc + n] = val;
                    if (C16) C16[(size_t)m * ldc + n] = (_Float16)val;
                }
            }
}

// ---------------------------------------------------------------------------
// Depthwise conv over seq axis, 'same' padding; output transposed [B,S,HID].
// ---------------------------------------------------------------------------
__global__ void dwconv_kernel(const float* __restrict__ key,
                              const float* __restrict__ dw_w,
                              const float* __restrict__ dw_b,
                              float* __restrict__ dwT)
{
    size_t i = (size_t)blockIdx.x * blockDim.x + threadIdx.x;
    if (i >= (size_t)MROWS * HIDc) return;
    int ch = (int)(i % HIDc);
    int s  = (int)((i / HIDc) % Ssz);
    int b  = (int)(i / ((size_t)HIDc * Ssz));
    float acc = dw_b[ch];
    #pragma unroll
    for (int t = 0; t < Kc; ++t) {
        int ss = s + t - Kc / 2;
        if (ss >= 0 && ss < Ssz)
            acc += key[((size_t)b * Ssz + ss) * HIDc + ch] * dw_w[ch * Kc + t];
    }
    dwT[i] = acc;
}

// ---------------------------------------------------------------------------
// Fused attention: per block = (b, h, 16 query rows), 128 threads / 4 waves.
// ---------------------------------------------------------------------------
__global__ __launch_bounds__(128) void attn_kernel(
    const _Float16* __restrict__ q16, const _Float16* __restrict__ k16,
    const _Float16* __restrict__ v16, const int* __restrict__ mask,
    float* __restrict__ attn_wts, float* __restrict__ attn_out)
{
    __shared__ float     sc[16][Ssz];        // 64 KB raw scores (f32)
    __shared__ _Float16  p16[16][Ssz + 8];   // 32 KB f16 probs (A fragments)
    __shared__ _Float16  Vs[32][72];         // 4.5 KB staged V tile

    const int tid  = threadIdx.x;
    const int wave = tid >> 5;
    const int lane = tid & 31;
    const int hf   = lane >> 4;
    const int l16  = lane & 15;

    const int qb = blockIdx.x & 63;
    const int h  = (blockIdx.x >> 6) % NHc;
    const int b  = blockIdx.x / (64 * NHc);
    const int q0 = qb * 16;

    v16h aq[2];
    {
        const _Float16* qrow = q16 + ((size_t)b * Ssz + q0 + l16) * RHc + h * HDc;
        #pragma unroll
        for (int t = 0; t < 2; ++t) {
            v8h lo = *(const v8h*)(qrow + t * 32 + hf * 16);
            v8h hi = *(const v8h*)(qrow + t * 32 + hf * 16 + 8);
            #pragma unroll
            for (int j = 0; j < 8; ++j) { aq[t][j] = lo[j]; aq[t][8 + j] = hi[j]; }
        }
    }

    // ---- phase 1: scores = q @ k^T / sqrt(HD) + mask -> LDS ----
    for (int kb = wave; kb < Ssz / 16; kb += 4) {
        v8f acc = {};
        #pragma unroll
        for (int t = 0; t < 2; ++t) {
            const _Float16* krow = k16 + ((size_t)b * Ssz + kb * 16 + l16) * RHc
                                       + h * HDc + t * 32 + hf * 16;
            v8h lo = *(const v8h*)(krow);
            v8h hi = *(const v8h*)(krow + 8);
            v16h bf;
            #pragma unroll
            for (int j = 0; j < 8; ++j) { bf[j] = lo[j]; bf[8 + j] = hi[j]; }
            acc = __builtin_amdgcn_wmma_f32_16x16x32_f16(false, aq[t], false, bf,
                                                         (short)0, acc, false, false);
        }
        const int col = kb * 16 + l16;
        const float msk = (mask[b * Ssz + col] == 0) ? -INFINITY : 0.f;
        #pragma unroll
        for (int r = 0; r < 8; ++r)
            sc[hf * 8 + r][col] = acc[r] * 0.125f + msk;
    }
    __syncthreads();

    // ---- phase 2: row softmax; write attn_wts (output #1) + f16 probs ----
    for (int r = wave; r < 16; r += 4) {
        float mx = -INFINITY;
        for (int c = lane; c < Ssz; c += 32) mx = fmaxf(mx, sc[r][c]);
        #pragma unroll
        for (int o = 16; o > 0; o >>= 1) mx = fmaxf(mx, __shfl_xor(mx, o, 32));
        float sum = 0.f;
        for (int c = lane; c < Ssz; c += 32) sum += __expf(sc[r][c] - mx);
        #pragma unroll
        for (int o = 16; o > 0; o >>= 1) sum += __shfl_xor(sum, o, 32);
        const float inv = 1.f / sum;
        float* dst = attn_wts + (((size_t)b * NHc + h) * Ssz + (q0 + r)) * Ssz;
        for (int c = lane; c < Ssz; c += 32) {
            float p = __expf(sc[r][c] - mx) * inv;
            p16[r][c] = (_Float16)p;
            dst[c] = p;
        }
    }
    __syncthreads();

    // ---- phase 3: out = probs @ V ; V tiles staged in LDS ----
    v8f acc = {};
    const int vr = tid >> 2;
    const int vc = (tid & 3) * 16;
    for (int kk = 0; kk < Ssz; kk += 32) {
        const _Float16* vsrc = v16 + ((size_t)b * Ssz + kk + vr) * RHc + h * HDc + vc;
        *(v8h*)&Vs[vr][vc]     = *(const v8h*)(vsrc);
        *(v8h*)&Vs[vr][vc + 8] = *(const v8h*)(vsrc + 8);
        __syncthreads();

        v16h a, bf;
        v8h lo = *(const v8h*)&p16[l16][kk + hf * 16];
        v8h hi = *(const v8h*)&p16[l16][kk + hf * 16 + 8];
        #pragma unroll
        for (int j = 0; j < 8; ++j) { a[j] = lo[j]; a[8 + j] = hi[j]; }
        #pragma unroll
        for (int j = 0; j < 16; ++j)
            bf[j] = Vs[hf * 16 + j][wave * 16 + l16];
        acc = __builtin_amdgcn_wmma_f32_16x16x32_f16(false, a, false, bf,
                                                     (short)0, acc, false, false);
        __syncthreads();
    }
    #pragma unroll
    for (int r = 0; r < 8; ++r) {
        int m = q0 + hf * 8 + r;
        attn_out[((size_t)b * Ssz + m) * RHc + h * HDc + wave * 16 + l16] = acc[r];
    }
}

// ---------------------------------------------------------------------------
__global__ void mul_kernel(const float* __restrict__ a, const float* __restrict__ b,
                           float* __restrict__ o, size_t n)
{
    size_t i = (size_t)blockIdx.x * blockDim.x + threadIdx.x;
    if (i < n) o[i] = a[i] * b[i];
}

// ---------------------------------------------------------------------------
__global__ void dynconv_kernel(const float* __restrict__ v,
                               const float* __restrict__ logits,
                               float* __restrict__ dyn_out)
{
    size_t i = (size_t)blockIdx.x * blockDim.x + threadIdx.x;
    if (i >= (size_t)MROWS * RHc) return;
    int hd = (int)(i % RHc);
    int s  = (int)((i / RHc) % Ssz);
    int b  = (int)(i / ((size_t)RHc * Ssz));
    int h  = hd / HDc;
    const float* lg = logits + ((size_t)b * Ssz + s) * (Kc * NHc) + h * Kc;
    float mx = lg[0];
    #pragma unroll
    for (int t = 1; t < Kc; ++t) mx = fmaxf(mx, lg[t]);
    float w[Kc], sum = 0.f;
    #pragma unroll
    for (int t = 0; t < Kc; ++t) { w[t] = __expf(lg[t] - mx); sum += w[t]; }
    float inv = 1.f / sum;
    float acc = 0.f;
    #pragma unroll
    for (int t = 0; t < Kc; ++t) {
        int ss = s + t - Kc / 2;
        if (ss >= 0 && ss < Ssz)
            acc += v[((size_t)b * Ssz + ss) * RHc + hd] * (w[t] * inv);
    }
    dyn_out[i] = acc;
}

// ---------------------------------------------------------------------------
extern "C" void kernel_launch(void* const* d_in, const int* in_sizes, int n_in,
                              void* d_out, int out_size, void* d_ws, size_t ws_size,
                              hipStream_t stream)
{
    const float* query = (const float*)d_in[0];
    const int*   mask  = (const int*)d_in[1];
    const float* Wq  = (const float*)d_in[2];  const float* bq  = (const float*)d_in[3];
    const float* Wk  = (const float*)d_in[4];  const float* bk  = (const float*)d_in[5];
    const float* Wv  = (const float*)d_in[6];  const float* bv  = (const float*)d_in[7];
    const float* dw_w = (const float*)d_in[8]; const float* dw_b = (const float*)d_in[9];
    const float* pw_w = (const float*)d_in[10];const float* pw_b = (const float*)d_in[11];
    const float* Wsa = (const float*)d_in[12]; const float* bsa = (const float*)d_in[13];
    const float* Wdk = (const float*)d_in[14]; const float* bdk = (const float*)d_in[15];
    const float* Wda = (const float*)d_in[16]; const float* bda = (const float*)d_in[17];
    const float* keyp = (const float*)d_in[18];
    const float* valp = (const float*)d_in[19];

    float* ws   = (float*)d_ws;
    float* q    = ws;                               // 8192*384
    float* kprj = q    + (size_t)MROWS * RHc;
    float* vprj = kprj + (size_t)MROWS * RHc;
    float* dwT  = vprj + (size_t)MROWS * RHc;       // 8192*768
    float* ksf  = dwT  + (size_t)MROWS * HIDc;
    float* atto = ksf  + (size_t)MROWS * RHc;
    float* qks  = atto + (size_t)MROWS * RHc;
    float* dynl = qks  + (size_t)MROWS * RHc;       // 8192*54
    float* dyno = dynl + (size_t)MROWS * (Kc * NHc);

    _Float16* q16   = (_Float16*)(dyno + (size_t)MROWS * RHc);
    _Float16* k16   = q16  + (size_t)MROWS * RHc;
    _Float16* v16   = k16  + (size_t)MROWS * RHc;
    _Float16* Wq16  = v16  + (size_t)MROWS * RHc;
    _Float16* Wk16  = Wq16 + (size_t)RHc * HIDc;
    _Float16* Wv16  = Wk16 + (size_t)RHc * HIDc;
    _Float16* pw16  = Wv16 + (size_t)RHc * HIDc;
    _Float16* Wsa16 = pw16 + (size_t)RHc * HIDc;
    _Float16* Wda16 = Wsa16 + (size_t)RHc * RHc;
    _Float16* Wdk16 = Wda16 + (size_t)RHc * RHc;

    float* attn_wts = (float*)d_out;
    float* outc = attn_wts + (size_t)Bsz * NHc * Ssz * Ssz;

    #define CVT(W, O, N, K, T) \
        convert_wt<<<((N) * (K) + 255) / 256, 256, 0, stream>>>(W, O, N, K, T)
    CVT(Wq,   Wq16,  RHc, HIDc, 1);
    CVT(Wk,   Wk16,  RHc, HIDc, 1);
    CVT(Wv,   Wv16,  RHc, HIDc, 1);
    CVT(pw_w, pw16,  RHc, HIDc, 0);
    CVT(Wsa,  Wsa16, RHc, RHc,  1);
    CVT(Wdk,  Wdk16, Kc * NHc, RHc, 1);
    CVT(Wda,  Wda16, RHc, RHc,  1);
    #undef CVT

    dim3 gP(RHc / 64, MROWS / 64);        // 6 x 128

    gemm_wmma<false><<<gP, 128, 0, stream>>>(query, HIDc, Wq16, HIDc, bq, q,    RHc, q16, MROWS, RHc, HIDc);
    gemm_wmma<false><<<gP, 128, 0, stream>>>(keyp,  HIDc, Wk16, HIDc, bk, kprj, RHc, k16, MROWS, RHc, HIDc);
    gemm_wmma<false><<<gP, 128, 0, stream>>>(valp,  HIDc, Wv16, HIDc, bv, vprj, RHc, v16, MROWS, RHc, HIDc);

    dwconv_kernel<<<(int)(((size_t)MROWS * HIDc + 255) / 256), 256, 0, stream>>>(keyp, dw_w, dw_b, dwT);
    gemm_wmma<false><<<gP, 128, 0, stream>>>(dwT, HIDc, pw16, HIDc, pw_b, ksf, RHc, nullptr, MROWS, RHc, HIDc);

    attn_kernel<<<Bsz * NHc * (Ssz / 16), 128, 0, stream>>>(q16, k16, v16, mask, attn_wts, atto);

    mul_kernel<<<(int)(((size_t)MROWS * RHc + 255) / 256), 256, 0, stream>>>(q, ksf, qks, (size_t)MROWS * RHc);
    dim3 gL(1, MROWS / 64);
    gemm_wmma<true><<<gL, 128, 0, stream>>>(qks, RHc, Wdk16, RHc, bdk, dynl, Kc * NHc, nullptr, MROWS, Kc * NHc, RHc);
    dynconv_kernel<<<(int)(((size_t)MROWS * RHc + 255) / 256), 256, 0, stream>>>(vprj, dynl, dyno);

    gemm_wmma<false><<<gP, 128, 0, stream>>>(atto, RHc, Wsa16, RHc, bsa, outc,       HIDc, nullptr, MROWS, RHc, RHc);
    gemm_wmma<false><<<gP, 128, 0, stream>>>(dyno, RHc, Wda16, RHc, bda, outc + RHc, HIDc, nullptr, MROWS, RHc, RHc);
}
